// FC_large_11879879544390
// MI455X (gfx1250) — 8-bit matrix operand order repaired and checked by reading, compiled, not run on hardware
//
#include <hip/hip_runtime.h>
#include <hip/hip_bf16.h>


typedef __attribute__((ext_vector_type(16))) int   v16i;
typedef __attribute__((ext_vector_type(8)))  float v8f;
typedef __attribute__((ext_vector_type(4)))  int   v4i;
typedef __attribute__((ext_vector_type(2)))  int   v2i;

#define FP8_P1 0x38u
#define FP8_M1 0xB8u

static __device__ __forceinline__ unsigned int sign4(float a, float b, float c, float d) {
  unsigned int r = (a >= 0.f ? FP8_P1 : FP8_M1);
  r |= (b >= 0.f ? FP8_P1 : FP8_M1) << 8;
  r |= (c >= 0.f ? FP8_P1 : FP8_M1) << 16;
  r |= (d >= 0.f ? FP8_P1 : FP8_M1) << 24;
  return r;
}

static __device__ __forceinline__ unsigned lds_off(const void* p) {
  return (unsigned)(unsigned long long)p;
}

static __device__ __forceinline__ void async_b128(unsigned ldsaddr, const void* gaddr) {
  asm volatile("global_load_async_to_lds_b128 %0, %1, off"
               :: "v"(ldsaddr), "v"(gaddr)
               : "memory");
}

__global__ void k_pack_x(const float* __restrict__ x, unsigned int* __restrict__ xq, int M) {
  int idx = blockIdx.x * blockDim.x + threadIdx.x;
  int total = M * 192;
  if (idx >= total) return;
  int r = idx / 192, c4 = idx - r * 192;
  float4 v = *((const float4*)(x + (size_t)r * 784) + c4);
  xq[idx] = sign4(v.x, v.y, v.z, v.w);
}

__global__ void k_pack_w(const float* __restrict__ w, unsigned int* __restrict__ wq, int total4) {
  int idx = blockIdx.x * blockDim.x + threadIdx.x;
  if (idx >= total4) return;
  float4 v = ((const float4*)w)[idx];
  wq[idx] = sign4(v.x, v.y, v.z, v.w);
}

__global__ void k_pack_w4(const float* __restrict__ w, unsigned int* __restrict__ wq) {
  int idx = blockIdx.x * blockDim.x + threadIdx.x;
  if (idx >= 16 * 1024) return;
  int r = idx >> 10;
  if (r < 10) {
    float4 v = ((const float4*)w)[idx];
    wq[idx] = sign4(v.x, v.y, v.z, v.w);
  } else {
    wq[idx] = 0u;
  }
}

#define LDS_STRIDE 144

template <int NT>
__global__ __launch_bounds__(256) void k_gemm_fp8(
    const unsigned char* __restrict__ A,
    const unsigned char* __restrict__ B,
    float* __restrict__ C,
    int N, int K) {
  __shared__ unsigned char sA[2][128 * LDS_STRIDE];
  __shared__ unsigned char sB[2][NT * 16 * LDS_STRIDE];

  const int tid   = threadIdx.x;
  const int lane  = tid & 31;
  const int wave  = tid >> 5;
  const int m0    = blockIdx.x * 128;
  const int n0    = blockIdx.y * (NT * 16);
  const int lrow  = lane & 15;
  const int hi    = lane >> 4;
  const int NK    = K >> 7;

  auto stage = [&](int buf, int ks) {
    const int k0 = ks << 7;
#pragma unroll
    for (int i = 0; i < 4; ++i) {
      int ca  = tid + i * 256;
      int row = ca >> 3, cb = (ca & 7) * 16;
      async_b128(lds_off(&sA[buf][row * LDS_STRIDE + cb]),
                 A + (size_t)(m0 + row) * K + (k0 + cb));
    }
#pragma unroll
    for (int i = 0; i < 4; ++i) {
      int cbk = (tid + i * 256) & (NT * 128 - 1);
      int col = cbk >> 3, cb = (cbk & 7) * 16;
      async_b128(lds_off(&sB[buf][col * LDS_STRIDE + cb]),
                 B + (size_t)(n0 + col) * K + (k0 + cb));
    }
  };

  auto load_bfrag = [&](int buf, int t) -> v16i {
    const unsigned char* bT = &sB[buf][(t * 16 + lrow) * LDS_STRIDE + hi * 16];
    v16i bf;
#pragma unroll
    for (int j = 0; j < 4; ++j) {
      v4i c4 = *(const v4i*)(bT + j * 32);
      bf[4 * j]     = c4[0];
      bf[4 * j + 1] = c4[1];
      bf[4 * j + 2] = c4[2];
      bf[4 * j + 3] = c4[3];
    }
    return bf;
  };

  v8f acc[NT] = {};

  stage(0, 0);
  for (int kt = 0; kt < NK; ++kt) {
    const int cur = kt & 1;
    if (kt + 1 < NK) {
      stage(cur ^ 1, kt + 1);
      asm volatile("s_wait_asynccnt 0x8" ::: "memory");
    } else {
      asm volatile("s_wait_asynccnt 0x0" ::: "memory");
    }
    __syncthreads();

    const unsigned char* aT = &sA[cur][(wave * 16 + lrow) * LDS_STRIDE + hi * 16];
    v16i af;
#pragma unroll
    for (int j = 0; j < 8; ++j) {
      v2i c2 = *(const v2i*)(aT + (j >> 1) * 32 + (j & 1) * 8);
      af[2 * j]     = c2[0];
      af[2 * j + 1] = c2[1];
    }

    v16i bf_cur = load_bfrag(cur, 0);
#pragma unroll
    for (int t = 0; t < NT; ++t) {
      v16i bf_nxt;
      if (t + 1 < NT) bf_nxt = load_bfrag(cur, t + 1);
      acc[t] = __builtin_amdgcn_wmma_f32_16x16x128_fp8_fp8(
          af, bf_cur, (short)0, acc[t], false, false);
      if (t + 1 < NT) bf_cur = bf_nxt;
    }
    __syncthreads();
  }

  const int crow = m0 + wave * 16 + hi * 8;
#pragma unroll
  for (int t = 0; t < NT; ++t) {
    float* cp = C + (size_t)crow * N + (n0 + t * 16 + lrow);
#pragma unroll
    for (int i = 0; i < 8; ++i) cp[(size_t)i * N] = acc[t][i];
  }
}

__global__ void k_colpart(const float* __restrict__ Y, float* __restrict__ part,
                          int N, int rowsPerChunk) {
  int c = blockIdx.x * blockDim.x + threadIdx.x;
  if (c >= N) return;
  int r0 = blockIdx.y * rowsPerChunk;
  float s = 0.f, s2 = 0.f;
  for (int r = 0; r < rowsPerChunk; ++r) {
    float v = Y[(size_t)(r0 + r) * N + c];
    s += v;
    s2 = fmaf(v, v, s2);
  }
  int RS = gridDim.y;
  part[(size_t)blockIdx.y * N + c]        = s;
  part[(size_t)(RS + blockIdx.y) * N + c] = s2;
}

__global__ void k_colfin(const float* __restrict__ part, const float* __restrict__ gamma,
                         const float* __restrict__ beta, float* __restrict__ ss,
                         int N, int NG, int RS, float invM) {
  int c = blockIdx.x * blockDim.x + threadIdx.x;
  if (c >= N) return;
  float s = 0.f, s2 = 0.f;
  for (int i = 0; i < RS; ++i) {
    s  += part[(size_t)i * N + c];
    s2 += part[(size_t)(RS + i) * N + c];
  }
  float mu  = s * invM;
  float var = fmaxf(s2 * invM - mu * mu, 0.f);
  float g = (c < NG) ? gamma[c] : 1.f;
  float b = (c < NG) ? beta[c]  : 0.f;
  float sc = g * rsqrtf(var + 1e-5f);
  ss[c]     = sc;
  ss[N + c] = b - mu * sc;
}

__global__ void k_bn_sign(const float* __restrict__ Y, const float* __restrict__ ss,
                          unsigned int* __restrict__ Hq, int M, int N) {
  int idx = blockIdx.x * blockDim.x + threadIdx.x;
  int n4 = N >> 2;
  if (idx >= M * n4) return;
  int c4 = idx % n4;
  float4 v = ((const float4*)Y)[idx];
  int c = c4 << 2;
  const float* sc = ss;
  const float* sh = ss + N;
  float z0 = fmaf(v.x, sc[c + 0], sh[c + 0]);
  float z1 = fmaf(v.y, sc[c + 1], sh[c + 1]);
  float z2 = fmaf(v.z, sc[c + 2], sh[c + 2]);
  float z3 = fmaf(v.w, sc[c + 3], sh[c + 3]);
  Hq[idx] = sign4(z0, z1, z2, z3);
}

__global__ void k_lsm(const float* __restrict__ Y4, const float* __restrict__ ss4,
                      float* __restrict__ out, int M) {
  int r = blockIdx.x * blockDim.x + threadIdx.x;
  if (r >= M) return;
  float z[10];
  float mx = -1e30f;
#pragma unroll
  for (int j = 0; j < 10; ++j) {
    float v = fmaf(Y4[(size_t)r * 16 + j], ss4[j], ss4[16 + j]);
    z[j] = v;
    mx = fmaxf(mx, v);
  }
  float s = 0.f;
#pragma unroll
  for (int j = 0; j < 10; ++j) s += expf(z[j] - mx);
  float l = mx + logf(s);
#pragma unroll
  for (int j = 0; j < 10; ++j) out[(size_t)r * 10 + j] = z[j] - l;
}

extern "C" void kernel_launch(void* const* d_in, const int* in_sizes, int n_in,
                              void* d_out, int out_size, void* d_ws, size_t ws_size,
                              hipStream_t stream) {
  const float* x  = (const float*)d_in[0];
  const float* w1 = (const float*)d_in[1];
  const float* w2 = (const float*)d_in[2];
  const float* w3 = (const float*)d_in[3];
  const float* w4 = (const float*)d_in[4];
  const float* g1 = (const float*)d_in[5];  const float* b1 = (const float*)d_in[6];
  const float* g2 = (const float*)d_in[7];  const float* b2 = (const float*)d_in[8];
  const float* g3 = (const float*)d_in[9];  const float* b3 = (const float*)d_in[10];
  const float* g4 = (const float*)d_in[11]; const float* b4 = (const float*)d_in[12];
  float* out = (float*)d_out;

  const int Bsz = 16384, HID = 4096, IND = 768;
  const int RS = 128;
  const float invM = 1.0f / (float)Bsz;

  unsigned char* ws = (unsigned char*)d_ws;
  size_t off = 0;
  auto carve = [&](size_t bytes) -> void* {
    void* p = ws + off;
    off = (off + bytes + 255) & ~(size_t)255;
    return p;
  };
  unsigned char* Xq  = (unsigned char*)carve((size_t)Bsz * IND);
  unsigned char* W1q = (unsigned char*)carve((size_t)HID * IND);
  unsigned char* W2q = (unsigned char*)carve((size_t)HID * HID);
  unsigned char* W3q = (unsigned char*)carve((size_t)HID * HID);
  unsigned char* W4q = (unsigned char*)carve((size_t)16 * HID);
  float* Y    = (float*)carve((size_t)Bsz * HID * 4);
  unsigned char* Hq = (unsigned char*)carve((size_t)Bsz * HID);
  float* part = (float*)carve((size_t)2 * RS * HID * 4);
  float* ssb  = (float*)carve((size_t)2 * HID * 4);
  float* Y4   = (float*)carve((size_t)Bsz * 16 * 4);
  float* ss4  = (float*)carve((size_t)2 * 16 * 4);
  (void)ws_size; (void)in_sizes; (void)n_in; (void)out_size;

  k_pack_x<<<(Bsz * 192 + 255) / 256, 256, 0, stream>>>(x, (unsigned int*)Xq, Bsz);
  k_pack_w<<<(HID * IND / 4 + 255) / 256, 256, 0, stream>>>(w1, (unsigned int*)W1q, HID * IND / 4);
  k_pack_w<<<(HID * HID / 4 + 255) / 256, 256, 0, stream>>>(w2, (unsigned int*)W2q, HID * HID / 4);
  k_pack_w<<<(HID * HID / 4 + 255) / 256, 256, 0, stream>>>(w3, (unsigned int*)W3q, HID * HID / 4);
  k_pack_w4<<<(16 * 1024 + 255) / 256, 256, 0, stream>>>(w4, (unsigned int*)W4q);

  k_gemm_fp8<8><<<dim3(Bsz / 128, HID / 128), 256, 0, stream>>>(Xq, W1q, Y, HID, IND);
  k_colpart<<<dim3(HID / 256, RS), 256, 0, stream>>>(Y, part, HID, Bsz / RS);
  k_colfin<<<HID / 256, 256, 0, stream>>>(part, g1, b1, ssb, HID, HID, RS, invM);
  k_bn_sign<<<(Bsz * HID / 4 + 255) / 256, 256, 0, stream>>>(Y, ssb, (unsigned int*)Hq, Bsz, HID);

  k_gemm_fp8<8><<<dim3(Bsz / 128, HID / 128), 256, 0, stream>>>(Hq, W2q, Y, HID, HID);
  k_colpart<<<dim3(HID / 256, RS), 256, 0, stream>>>(Y, part, HID, Bsz / RS);
  k_colfin<<<HID / 256, 256, 0, stream>>>(part, g2, b2, ssb, HID, HID, RS, invM);
  k_bn_sign<<<(Bsz * HID / 4 + 255) / 256, 256, 0, stream>>>(Y, ssb, (unsigned int*)Hq, Bsz, HID);

  k_gemm_fp8<8><<<dim3(Bsz / 128, HID / 128), 256, 0, stream>>>(Hq, W3q, Y, HID, HID);
  k_colpart<<<dim3(HID / 256, RS), 256, 0, stream>>>(Y, part, HID, Bsz / RS);
  k_colfin<<<HID / 256, 256, 0, stream>>>(part, g3, b3, ssb, HID, HID, RS, invM);
  k_bn_sign<<<(Bsz * HID / 4 + 255) / 256, 256, 0, stream>>>(Y, ssb, (unsigned int*)Hq, Bsz, HID);

  k_gemm_fp8<1><<<dim3(Bsz / 128, 1), 256, 0, stream>>>(Hq, W4q, Y4, 16, HID);
  k_colpart<<<dim3(1, RS), 256, 0, stream>>>(Y4, part, 16, Bsz / RS);
  k_colfin<<<1, 256, 0, stream>>>(part, g4, b4, ss4, 16, 10, RS, invM);
  k_lsm<<<(Bsz + 255) / 256, 256, 0, stream>>>(Y4, ss4, out, Bsz);
}
